// KPConvBlock_26225070309987
// MI455X (gfx1250) — matrix operand order repaired and checked by reading, not compiled, not run on hardware
//
#include <hip/hip_runtime.h>
#include <math.h>

// ---------------- problem constants ----------------
#define M_PTS   50000
#define N_PTS   50000
#define H_NB    32
#define K_KP    15
#define C_IN    64
#define C_OUT   64
#define SIGMA_  0.05f
#define BN_EPS_ 1e-5f
#define SLOPE_  0.1f

#define QPG     32            // queries per workgroup (2 WMMA row-tiles)
#define KC_PAD  1024          // reduction dim: kc' = c*16 + k (k padded to 16)
#define NSTEP   (KC_PAD / 32) // 32 K-steps
#define WF_STRIDE 1024        // wf row stride in halves: [q][c(64)][k(16)]
#define WSWZ_HALVES (NSTEP * 4 * 32 * 16)   // 65536 f16 = 131072 B

// LDS layout (bytes)
#define LDS_COORDS 0
#define LDS_INDS   16384
#define LDS_NF     20480                     // 4 waves x 2 bufs x 32x64 f32 = 64KB
#define LDS_WF     86016                     // 32 x 1024 f16 = 64KB
#define LDS_TOTAL  151552

typedef _Float16 half_t;
typedef __attribute__((ext_vector_type(16))) _Float16 v16h;
typedef __attribute__((ext_vector_type(8)))  _Float16 v8h;
typedef __attribute__((ext_vector_type(8)))  float    v8f;
typedef int v4i_t __attribute__((ext_vector_type(4)));
typedef __attribute__((address_space(1))) v4i_t* gv4_ptr;   // global (AS1)
typedef __attribute__((address_space(3))) v4i_t* lv4_ptr;   // LDS (AS3)

#if __has_builtin(__builtin_amdgcn_global_load_async_to_lds_b128)
#define HAVE_ASYNC_LDS 1
#else
#define HAVE_ASYNC_LDS 0
#endif

template <int N>
__device__ __forceinline__ void wait_async_le() {
#if HAVE_ASYNC_LDS
#if __has_builtin(__builtin_amdgcn_s_wait_asynccnt)
    __builtin_amdgcn_s_wait_asynccnt(N);
    asm volatile("" ::: "memory");
#else
    asm volatile("s_wait_asynccnt %0" :: "i"(N) : "memory");
#endif
#else
    asm volatile("s_wait_dscnt 0" ::: "memory");
#endif
}

// Copy one 256B feature row (64 f32) global -> LDS for this lane (async DMA).
__device__ __forceinline__ void gather_row(const float* __restrict__ fr, float* lr) {
#if HAVE_ASYNC_LDS
#pragma unroll
    for (int c0 = 0; c0 < C_IN; c0 += 4)
        __builtin_amdgcn_global_load_async_to_lds_b128(
            (gv4_ptr)(fr + c0), (lv4_ptr)(lr + c0), 0, 0);
#else
#pragma unroll
    for (int c0 = 0; c0 < C_IN; c0 += 4)
        *(float4*)(lr + c0) = *(const float4*)(fr + c0);
#endif
}

// ---------------------------------------------------------------------------
// K0: zero BN stats; convert+swizzle weights into per-lane f16 B-fragments
//     over the kc' = c*16 + k (k padded to 16, zero-filled) reduction order.
//     flat n = ((s*4 + t)*32 + l)*16 + 2j + p
//     row  kc' = s*32 + (l>=16 ? 16 : 0) + 2j + p ;  col d = t*16 + (l&15)
// ---------------------------------------------------------------------------
__global__ __launch_bounds__(256) void kpconv_prep(
    const float* __restrict__ weights, half_t* __restrict__ wswz,
    float* __restrict__ stats)
{
    int n = blockIdx.x * 256 + threadIdx.x;
    if (n < 128) stats[n] = 0.0f;
    if (n < WSWZ_HALVES) {
        int p = n & 1;
        int j = (n >> 1) & 7;
        int l = (n >> 4) & 31;
        int t = (n >> 9) & 3;
        int s = n >> 11;
        int kcp = s * 32 + ((l >= 16) ? 16 : 0) + 2 * j + p;  // < 1024
        int k = kcp & 15;
        int c = kcp >> 4;
        int d = t * 16 + (l & 15);
        float v = (k < K_KP) ? weights[(k * C_IN + c) * C_OUT + d] : 0.0f;
        wswz[n] = (half_t)v;
    }
}

// ---------------------------------------------------------------------------
// K1: main KPConv kernel. 1 workgroup = 32 queries, 128 threads (4 waves).
// ---------------------------------------------------------------------------
__global__ __launch_bounds__(128) void kpconv_main(
    const float* __restrict__ q_pts, const float* __restrict__ s_pts,
    const float* __restrict__ s_feats, const int* __restrict__ nb_inds,
    const float* __restrict__ kpts, const half_t* __restrict__ wswz,
    float* __restrict__ out)
{
    extern __shared__ char smem[];
    float4* coords = (float4*)(smem + LDS_COORDS);     // [32][32] float4
    int*    indsh  = (int*)(smem + LDS_INDS);          // [32][32]
    float*  nfbuf  = (float*)(smem + LDS_NF);          // [4 waves][2][32][64] f32
    half_t* wf     = (half_t*)(smem + LDS_WF);         // [32][64c][16k] f16

    const int tid   = threadIdx.x;
    const int lane  = tid & 31;
    const int wave  = tid >> 5;
    const int group = blockIdx.x;
    const int hi16  = (lane >= 16);

    // ---- stage: centered neighbor coords + neighbor indices (cooperative) --
    for (int idx = tid; idx < QPG * H_NB; idx += 128) {
        int q = idx >> 5, h = idx & 31;
        int m = group * QPG + q; if (m >= M_PTS) m = M_PTS - 1;
        int nb = nb_inds[m * H_NB + h];
        indsh[idx] = nb;
        float qx = q_pts[m * 3 + 0], qy = q_pts[m * 3 + 1], qz = q_pts[m * 3 + 2];
        const float* sp = s_pts + (size_t)nb * 3;
        coords[idx] = make_float4(sp[0] - qx, sp[1] - qy, sp[2] - qz, 0.0f);
    }
    __syncthreads();

    // per-lane kernel point.  Pad row k=15 uses a far fake point so its
    // influence clamps to exactly zero (no per-element cndmask; D row 15 = 0).
    const int krow = lane & 15;
    float kpx = 1.0e3f, kpy = 1.0e3f, kpz = 1.0e3f;
    if (krow < K_KP) { kpx = kpts[krow*3]; kpy = kpts[krow*3+1]; kpz = kpts[krow*3+2]; }

    // ---- phase 1: per wave, 8 queries: wf[q] = w^T (16x32) x nf (32x64) ----
    // Double-buffered async gather: issue q+1 while computing q.
    float* nfw0 = nfbuf + wave * 2 * (H_NB * C_IN);
    float* nfw1 = nfw0 + H_NB * C_IN;

    {
        int q0 = wave * 8;
        int nb = indsh[q0 * H_NB + lane];
        gather_row(s_feats + (size_t)nb * C_IN, nfw0 + lane * C_IN);
    }

    for (int qi = 0; qi < 8; ++qi) {
        const int q = wave * 8 + qi;
        float* nfw = (qi & 1) ? nfw1 : nfw0;

        if (qi < 7) {
            int nb = indsh[(q + 1) * H_NB + lane];
            gather_row(s_feats + (size_t)nb * C_IN,
                       ((qi & 1) ? nfw0 : nfw1) + lane * C_IN);
            wait_async_le<16>();    // previous batch (16 async b128) done, in order
        } else {
            wait_async_le<0>();
        }

        // A fragment: influence w[k = lane&15][h] computed in registers.
        // A 16x32 f16 layout: VGPR j holds K = (j>>2)*16 + (hi16?8:0) + (j&3)*2 (+1)
        v16h afrag;
#pragma unroll
        for (int j = 0; j < 8; ++j) {
            int hbase = (j >> 2) * 16 + (hi16 ? 8 : 0) + (j & 3) * 2;
#pragma unroll
            for (int p = 0; p < 2; ++p) {
                float4 nc = coords[q * H_NB + hbase + p];
                float dx = nc.x - kpx, dy = nc.y - kpy, dz = nc.z - kpz;
                float dd = __builtin_amdgcn_sqrtf(dx*dx + dy*dy + dz*dz);
                float w  = 1.0f - dd * (1.0f / SIGMA_);
                w = (w > 0.0f) ? w : 0.0f;
                afrag[2*j + p] = (half_t)w;
            }
        }

        // 4 C-tiles: B from LDS nf f32 (32x16 slice, cvt to f16), WMMA,
        // D -> wf in [q][c][k16] order: lane's 8 rows are contiguous -> 1 b128.
#pragma unroll
        for (int t = 0; t < 4; ++t) {
            const int c = t * 16 + (lane & 15);
            v16h bfrag;
#pragma unroll
            for (int j = 0; j < 8; ++j) {
                int h0 = (j >> 2) * 16 + (hi16 ? 8 : 0) + (j & 3) * 2;
                bfrag[2*j]     = (half_t)nfw[h0 * C_IN + c];
                bfrag[2*j + 1] = (half_t)nfw[(h0 + 1) * C_IN + c];
            }
            v8f acc = {};
            acc = __builtin_amdgcn_wmma_f32_16x16x32_f16(
                false, afrag, false, bfrag, (short)0, acc, false, false);
            v8h dpack;
#pragma unroll
            for (int r = 0; r < 8; ++r) dpack[r] = (half_t)acc[r];
            *(v8h*)(wf + q * WF_STRIDE + c * 16 + hi16 * 8) = dpack;
        }
    }
    __syncthreads();

    // ---- phase 2: out(16q x 64) = wf_flat(16 x 1024) x Wswz(1024 x 64) -----
    const int qt  = wave >> 1;
    const int ct0 = (wave & 1) * 2;
    v8f acc0 = {}, acc1 = {};
    const half_t* wfrow = wf + (qt * 16 + (lane & 15)) * WF_STRIDE;

    for (int s = 0; s < NSTEP; ++s) {
        v16h afrag;
#pragma unroll
        for (int j = 0; j < 8; ++j) {
            int kc0 = s * 32 + (hi16 ? 16 : 0) + 2 * j;
            afrag[2*j]     = wfrow[kc0];
            afrag[2*j + 1] = wfrow[kc0 + 1];
        }
        v16h b0 = *(const v16h*)(wswz + (((s * 4 + ct0    ) * 32) + lane) * 16);
        v16h b1 = *(const v16h*)(wswz + (((s * 4 + ct0 + 1) * 32) + lane) * 16);
        acc0 = __builtin_amdgcn_wmma_f32_16x16x32_f16(
            false, afrag, false, b0, (short)0, acc0, false, false);
        acc1 = __builtin_amdgcn_wmma_f32_16x16x32_f16(
            false, afrag, false, b1, (short)0, acc1, false, false);
    }

#pragma unroll
    for (int r = 0; r < 8; ++r) {
        int qrow = qt * 16 + r + (hi16 ? 8 : 0);
        int m = group * QPG + qrow;
        if (m < M_PTS) {
            int d0 = ct0 * 16 + (lane & 15);
            out[(size_t)m * C_OUT + d0]      = acc0[r];
            out[(size_t)m * C_OUT + d0 + 16] = acc1[r];
        }
    }
}

// ---------------------------------------------------------------------------
// K2: per-channel sum / sumsq over M rows (BN training stats)
// ---------------------------------------------------------------------------
__global__ __launch_bounds__(256) void kpconv_stats(
    const float* __restrict__ out, float* __restrict__ stats)
{
    __shared__ float ssum[256], ssq[256];
    const int tid = threadIdx.x;
    const int c   = tid & 63;
    float s = 0.f, s2 = 0.f;
    for (int m = blockIdx.x * 4 + (tid >> 6); m < M_PTS; m += gridDim.x * 4) {
        float v = out[(size_t)m * C_OUT + c];
        s += v; s2 += v * v;
    }
    ssum[tid] = s; ssq[tid] = s2;
    __syncthreads();
    if (tid < 64) {
        float a = ssum[tid] + ssum[tid+64] + ssum[tid+128] + ssum[tid+192];
        float b = ssq[tid]  + ssq[tid+64]  + ssq[tid+128]  + ssq[tid+192];
        atomicAdd(&stats[tid],      a);
        atomicAdd(&stats[64 + tid], b);
    }
}

// ---------------------------------------------------------------------------
// K3: BatchNorm (biased var) + affine + leaky relu, in place on d_out
// ---------------------------------------------------------------------------
__global__ __launch_bounds__(256) void kpconv_bn(
    float* __restrict__ out, const float* __restrict__ stats,
    const float* __restrict__ gamma, const float* __restrict__ beta)
{
    const float invM = 1.0f / (float)M_PTS;
    size_t total = (size_t)M_PTS * C_OUT;
    for (size_t i = (size_t)blockIdx.x * 256 + threadIdx.x; i < total;
         i += (size_t)gridDim.x * 256) {
        int c = (int)(i & 63);
        float mean = stats[c] * invM;
        float var  = stats[64 + c] * invM - mean * mean;
        float rs   = rsqrtf(var + BN_EPS_);
        float v = (out[i] - mean) * rs * gamma[c] + beta[c];
        out[i] = (v > 0.0f) ? v : v * SLOPE_;
    }
}

// ---------------------------------------------------------------------------
extern "C" void kernel_launch(void* const* d_in, const int* in_sizes, int n_in,
                              void* d_out, int out_size, void* d_ws, size_t ws_size,
                              hipStream_t stream)
{
    const float* q_pts   = (const float*)d_in[0];
    const float* s_pts   = (const float*)d_in[1];
    const float* s_feats = (const float*)d_in[2];
    const int*   nb_inds = (const int*)d_in[3];
    const float* kpts    = (const float*)d_in[4];
    const float* weights = (const float*)d_in[5];
    const float* gamma   = (const float*)d_in[6];
    const float* beta    = (const float*)d_in[7];
    float* out = (float*)d_out;

    half_t* wswz  = (half_t*)d_ws;                                  // 131072 B
    float*  stats = (float*)((char*)d_ws + WSWZ_HALVES * sizeof(half_t));

    kpconv_prep<<<(WSWZ_HALVES + 255) / 256, 256, 0, stream>>>(weights, wswz, stats);

    int ngroups = (M_PTS + QPG - 1) / QPG;
    kpconv_main<<<ngroups, 128, LDS_TOTAL, stream>>>(q_pts, s_pts, s_feats, nb_inds,
                                                     kpts, wswz, out);

    kpconv_stats<<<512, 256, 0, stream>>>(out, stats);
    kpconv_bn<<<1024, 256, 0, stream>>>(out, stats, gamma, beta);
}
